// SelfAttention_58841051955817
// MI455X (gfx1250) — matrix operand order repaired and checked by reading, not compiled, not run on hardware
//
#include <hip/hip_runtime.h>

// ---------------------------------------------------------------------------
// Causal self-attention for MI455X (gfx1250, wave32, WMMA + TDM).
//   1) convert x, Wq/Wk/Wv  f32 -> bf16
//   2) QKV GEMM (bf16 WMMA): Q (pre-scaled 1/sqrt(d)), K, and V^T
//   3) S = Q @ K^T per batch, skipping tiles above the causal diagonal
//   4) causal row softmax -> P (bf16, zeros above diagonal)
//   5) O = P @ V (WMMA, K truncated at the diagonal), f32 out
// GEMM core: 128(M) x 128(N) block tile; the shared B panel (32K x 128N bf16)
// is DMA'd into LDS by the Tensor Data Mover (tensor_load_to_lds, double
// buffered, s_wait_tensorcnt), TDM-padded to an 80B pitch so ds_load_b128
// fragment reads are bank-conflict-free. The K loop is manually unrolled 2x
// (structural buffer/A-fragment alternation -> no rotation copies), and B
// fragments use a software-pipelined rotation so ds_load latency overlaps
// WMMA execution instead of a full s_wait_dscnt before every matrix op.
// ---------------------------------------------------------------------------

typedef __bf16 bf16_t;
typedef __attribute__((ext_vector_type(16))) __bf16 v16bf;
typedef __attribute__((ext_vector_type(8)))  float  v8f;
typedef __attribute__((ext_vector_type(4)))  unsigned int u32x4;
typedef __attribute__((ext_vector_type(8)))  int i32x8;
typedef __attribute__((ext_vector_type(4)))  int i32x4;

#define EMBED  1024
#define SEQ    2048
#define NBATCH 4
#define NROWS  (NBATCH * SEQ)   // 8192

// LDS B tile: 128 columns x 32 K bf16, padded row pitch 40 elements (80 B)
#define TILE_N     128
#define TILE_PITCH 40
#define TILE_ELE   (TILE_N * TILE_PITCH)

#if __has_builtin(__builtin_amdgcn_tensor_load_to_lds)
#define USE_TDM 1
#else
#define USE_TDM 0
#endif

// -------------------------- small helpers ---------------------------------

__device__ __forceinline__ bf16_t f2bf(float f) {
  unsigned u = __builtin_bit_cast(unsigned, f);
  u += 0x7FFFu + ((u >> 16) & 1u);            // round-to-nearest-even
  unsigned short h = (unsigned short)(u >> 16);
  return __builtin_bit_cast(bf16_t, h);
}

union FragU {
  struct { uint4 a, b; } q;   // 32 bytes
  v16bf v;                    // 16 x bf16
};

// A fragment (16x32 MxK): lane<16 -> row M=lane, K={0..15};
// lane>=16 -> row M=lane-16, K={16..31}. Two b128 loads per lane.
__device__ __forceinline__ v16bf load_frag_a(const bf16_t* rowPtr, int k0, int hi16) {
  FragU u;
  u.q.a = *reinterpret_cast<const uint4*>(rowPtr + k0 + hi16 * 16);
  u.q.b = *reinterpret_cast<const uint4*>(rowPtr + k0 + hi16 * 16 + 8);
  return u.v;
}

// B fragment from the padded LDS tile: 32 contiguous bytes per lane.
__device__ __forceinline__ v16bf load_frag_b_lds(const bf16_t* tp) {
  const uint4* p = reinterpret_cast<const uint4*>(tp);
  FragU u;
  u.q.a = p[0];
  u.q.b = p[1];
  return u.v;
}

#if USE_TDM
// TDM load of a 32(K) x 128(N) bf16 tile (row stride = strideEl elements)
// into LDS at byte offset ldsOff, padding 4 DWORDs every 16 DWORDs
// (LDS row pitch 64B + 16B = 80B).
__device__ __forceinline__ void tdm_load_tile(
    const bf16_t* gsrc, unsigned ldsOff, unsigned strideEl)
{
  unsigned long long ga = (unsigned long long)(uintptr_t)gsrc;
  u32x4 g0;
  g0.x = 1u;                                      // count=1, user descriptor
  g0.y = ldsOff;                                  // LDS byte address
  g0.z = (unsigned)(ga & 0xFFFFFFFFu);            // global_addr[31:0]
  g0.w = (unsigned)(ga >> 32) | 0x80000000u;      // global_addr[56:32], type=2
  i32x8 g1;
  g1[0] = (int)((1u << 16)    // data_size = 2 bytes
              | (1u << 20)    // pad_enable
              | (3u << 22)    // pad_interval: 16 DWORDs
              | (3u << 25));  // pad_amount:   4 DWORDs
  g1[1] = (int)(32u << 16);              // tensor_dim0 = 32 (lo16)
  g1[2] = (int)((unsigned)TILE_N << 16); // dim0 hi = 0, tensor_dim1 (lo16)
  g1[3] = (int)(32u << 16);              // tensor_dim1 hi = 0, tile_dim0 = 32
  g1[4] = TILE_N;                        // tile_dim1 = 128, tile_dim2 = 0
  g1[5] = (int)strideEl;                 // tensor_dim0_stride lo32
  g1[6] = 0;
  g1[7] = 0;
  i32x4 z4 = {0, 0, 0, 0};
#if __clang_major__ >= 23
  i32x8 z8 = {0, 0, 0, 0, 0, 0, 0, 0};
  __builtin_amdgcn_tensor_load_to_lds(g0, g1, z4, z4, z8, 0);
#else
  __builtin_amdgcn_tensor_load_to_lds(g0, g1, z4, z4, 0);
#endif
}
#endif

// 8-deep WMMA burst over one staged LDS panel; B fragments are rotated so
// fragment t+1's ds_loads are in flight while WMMA t executes.
__device__ __forceinline__ void mma_burst(
    const bf16_t* tb, v16bf a, v8f acc[8])
{
  v16bf b_cur = load_frag_b_lds(tb);
#pragma unroll
  for (int t = 0; t < 8; ++t) {
    v16bf b_next = b_cur;
    if (t < 7) b_next = load_frag_b_lds(tb + (16 * (t + 1)) * TILE_PITCH);
    acc[t] = __builtin_amdgcn_wmma_f32_16x16x32_bf16(
        false, a, false, b_cur, (short)0, acc[t], false, false);
    b_cur = b_next;
  }
}

// Block-cooperative GEMM core. Block = 256 threads = 8 waves; block tile is
// 128(M) x 128(N); wave w owns rows [waveRowBase, +16) and all 128 N columns
// (8 v8f accumulators). Bblock points at the 128-row K-major B panel
// (C = A @ Bblock^T). kLim is block-uniform and a multiple of 64.
__device__ __forceinline__ void block_mma_128x128(
    const bf16_t* __restrict__ A, size_t lda,
    const bf16_t* __restrict__ Bblock, size_t ldb,
    int waveRowBase, int kLim, bf16_t* tile, v8f acc[8])
{
  const int wave = threadIdx.x >> 5;
  const int lane = threadIdx.x & 31;
  const int hi16 = lane >> 4;
  const int l15  = lane & 15;
  const bf16_t* aRow = A + (size_t)(waveRowBase + l15) * lda;
  const int nIter = kLim >> 5;                    // even (>= 4) for all users

  const bf16_t* tb0 = &tile[hi16 * 16 + l15 * TILE_PITCH];
  const bf16_t* tb1 = tb0 + TILE_ELE;

#if USE_TDM
  const unsigned lds0 = (unsigned)(uintptr_t)&tile[0];
  const unsigned lds1 = lds0 + TILE_ELE * 2;      // bytes

  if (wave == 0)
    tdm_load_tile(Bblock, lds0, (unsigned)ldb);

  v16bf a0 = load_frag_a(aRow, 0, hi16);
  v16bf a1 = a0;

  for (int it = 0; it < nIter; it += 2) {
    const int k0 = it << 5;
    // ---- phase 0: consume buf0 / a0, prefetch buf1 ----
    if (wave == 0) {
      tdm_load_tile(Bblock + (size_t)(k0 + 32), lds1, (unsigned)ldb);
      __builtin_amdgcn_s_wait_tensorcnt(1);       // buf0 landed
    }
    __syncthreads();
    a1 = load_frag_a(aRow, k0 + 32, hi16);
    mma_burst(tb0, a0, acc);
    __syncthreads();

    // ---- phase 1: consume buf1 / a1, prefetch buf0 ----
    if (wave == 0) {
      if (it + 2 < nIter) {
        tdm_load_tile(Bblock + (size_t)(k0 + 64), lds0, (unsigned)ldb);
        __builtin_amdgcn_s_wait_tensorcnt(1);     // buf1 landed
      } else {
        __builtin_amdgcn_s_wait_tensorcnt(0);
      }
    }
    __syncthreads();
    if (it + 2 < nIter) a0 = load_frag_a(aRow, k0 + 64, hi16);
    mma_burst(tb1, a1, acc);
    __syncthreads();
  }
#else
  // Fallback: cooperative global->LDS staging, single buffer.
  const int trow = threadIdx.x >> 1;              // 0..127 (tile column)
  const int tpart = (threadIdx.x & 1) * 2;        // 16B chunk pair
  for (int it = 0; it < nIter; ++it) {
    const int k0 = it << 5;
    __syncthreads();
#pragma unroll
    for (int c = 0; c < 2; ++c)
      *reinterpret_cast<uint4*>(&tile[trow * TILE_PITCH + (tpart + c) * 8]) =
          *reinterpret_cast<const uint4*>(Bblock + (size_t)trow * ldb + k0 + (tpart + c) * 8);
    __syncthreads();
    v16bf a = load_frag_a(aRow, k0, hi16);
    mma_burst(tb0, a, acc);
  }
#endif
}

// -------------------------- kernels ---------------------------------------

__global__ __launch_bounds__(256) void convert_f32_bf16(
    const float* __restrict__ in, bf16_t* __restrict__ out, int n)
{
  int i = (blockIdx.x * 256 + threadIdx.x) * 4;
  if (i >= n) return;
  float4 f = *reinterpret_cast<const float4*>(in + i);
  union { bf16_t b[4]; uint2 u; } pk;
  pk.b[0] = f2bf(f.x); pk.b[1] = f2bf(f.y);
  pk.b[2] = f2bf(f.z); pk.b[3] = f2bf(f.w);
  *reinterpret_cast<uint2*>(out + i) = pk.u;
}

// grid = (64, 24): x -> 128-row tiles; y -> {Q,K,V} x 8 128-col tiles.
__global__ __launch_bounds__(256) void qkv_gemm_kernel(
    const bf16_t* __restrict__ xbf, const bf16_t* __restrict__ wbf,
    const float* __restrict__ bq, const float* __restrict__ bk,
    const float* __restrict__ bv,
    bf16_t* __restrict__ qbf, bf16_t* __restrict__ kbf,
    bf16_t* __restrict__ vtbf)
{
  __shared__ bf16_t tile[2 * TILE_ELE];
  const int wave = threadIdx.x >> 5;
  const int lane = threadIdx.x & 31;
  const int rowBase = blockIdx.x * 128 + wave * 16;
  const int mat     = blockIdx.y >> 3;          // 0=Q, 1=K, 2=V
  const int colBase = (blockIdx.y & 7) * TILE_N;

  const bf16_t* W    = wbf + (size_t)mat * EMBED * EMBED;
  const float*  bias = (mat == 0) ? bq : (mat == 1) ? bk : bv;
  const float   scale = (mat == 0) ? 0.03125f : 1.0f;  // 1/sqrt(1024)

  v8f acc[8] = {};
  block_mma_128x128(xbf, EMBED, W + (size_t)colBase * EMBED, EMBED,
                    rowBase, EMBED, tile, acc);

  const int hi16 = lane >> 4;
#pragma unroll
  for (int t = 0; t < 8; ++t) {
    const int col = colBase + 16 * t + (lane & 15);
    const float bb = bias[col];
#pragma unroll
    for (int j = 0; j < 8; ++j) {
      const int row = rowBase + j + 8 * hi16;     // global row in [0, 8192)
      const float v = (acc[t][j] + bb) * scale;
      if (mat == 0) {
        qbf[(size_t)row * EMBED + col] = f2bf(v);
      } else if (mat == 1) {
        kbf[(size_t)row * EMBED + col] = f2bf(v);
      } else {
        const int b = row >> 11, s = row & (SEQ - 1);
        vtbf[((size_t)b * EMBED + col) * SEQ + s] = f2bf(v);
      }
    }
  }
}

// S = Q @ K^T per batch. grid = (16, 16, 4); above-diagonal tiles skipped.
__global__ __launch_bounds__(256) void scores_kernel(
    const bf16_t* __restrict__ qbf, const bf16_t* __restrict__ kbf,
    float* __restrict__ S)
{
  const int blockRow = blockIdx.x * 128;
  const int colBase  = blockIdx.y * TILE_N;
  if (colBase > blockRow + 127) return;           // fully masked tile

  __shared__ bf16_t tile[2 * TILE_ELE];
  const int b    = blockIdx.z;
  const int wave = threadIdx.x >> 5;
  const int lane = threadIdx.x & 31;
  const int rowBase = blockRow + wave * 16;

  const bf16_t* Q  = qbf + (size_t)b * SEQ * EMBED;
  const bf16_t* Kp = kbf + (size_t)b * SEQ * EMBED;
  float*        Sb = S   + (size_t)b * SEQ * SEQ;

  v8f acc[8] = {};
  block_mma_128x128(Q, EMBED, Kp + (size_t)colBase * EMBED, EMBED,
                    rowBase, EMBED, tile, acc);

  const int hi16 = lane >> 4;
#pragma unroll
  for (int t = 0; t < 8; ++t) {
    const int col = colBase + 16 * t + (lane & 15);
#pragma unroll
    for (int j = 0; j < 8; ++j) {
      const int row = rowBase + j + 8 * hi16;
      Sb[(size_t)row * SEQ + col] = acc[t][j];
    }
  }
}

__device__ __forceinline__ float block_reduce_max(float v) {
  __shared__ float s[8];
  const int lane = threadIdx.x & 31, wid = threadIdx.x >> 5;
#pragma unroll
  for (int off = 16; off > 0; off >>= 1)
    v = fmaxf(v, __shfl_xor(v, off, 32));
  if (lane == 0) s[wid] = v;
  __syncthreads();
  if (threadIdx.x == 0) {
    float r = s[0];
#pragma unroll
    for (int w = 1; w < 8; ++w) r = fmaxf(r, s[w]);
    s[0] = r;
  }
  __syncthreads();
  float r = s[0];
  __syncthreads();
  return r;
}

__device__ __forceinline__ float block_reduce_sum(float v) {
  __shared__ float s[8];
  const int lane = threadIdx.x & 31, wid = threadIdx.x >> 5;
#pragma unroll
  for (int off = 16; off > 0; off >>= 1)
    v += __shfl_xor(v, off, 32);
  if (lane == 0) s[wid] = v;
  __syncthreads();
  if (threadIdx.x == 0) {
    float r = s[0];
#pragma unroll
    for (int w = 1; w < 8; ++w) r += s[w];
    s[0] = r;
  }
  __syncthreads();
  float r = s[0];
  __syncthreads();
  return r;
}

// One block per (row, batch): causal softmax, P bf16 with zeros above diag.
__global__ __launch_bounds__(256) void softmax_kernel(
    const float* __restrict__ S, bf16_t* __restrict__ P)
{
  const int i = blockIdx.x, b = blockIdx.y;
  const float* srow = S + ((size_t)b * SEQ + i) * SEQ;
  bf16_t*      prow = P + ((size_t)b * SEQ + i) * SEQ;
  const int tid = threadIdx.x;
  const int n = i + 1;

  float m = -3.0e38f;
  for (int j = tid; j < n; j += 256) m = fmaxf(m, srow[j]);
  m = block_reduce_max(m);

  float l = 0.0f;
  for (int j = tid; j < n; j += 256) l += __expf(srow[j] - m);
  l = block_reduce_sum(l);
  const float inv = 1.0f / l;

  for (int j = tid; j < SEQ; j += 256) {
    const float p = (j < n) ? __expf(srow[j] - m) * inv : 0.0f;
    prow[j] = f2bf(p);
  }
}

// O = P @ V per batch. grid = (16, 8, 4). Block-uniform causal K limit.
__global__ __launch_bounds__(256) void pv_kernel(
    const bf16_t* __restrict__ P, const bf16_t* __restrict__ Vt,
    float* __restrict__ out)
{
  __shared__ bf16_t tile[2 * TILE_ELE];
  const int b    = blockIdx.z;
  const int wave = threadIdx.x >> 5;
  const int lane = threadIdx.x & 31;
  const int blockRow = blockIdx.x * 128;
  const int rowBase  = blockRow + wave * 16;
  const int colBase  = blockIdx.y * TILE_N;

  const bf16_t* Pb = P   + (size_t)b * SEQ * SEQ;
  const bf16_t* Vb = Vt  + (size_t)b * EMBED * SEQ;
  float*        Ob = out + (size_t)b * SEQ * EMBED;

  const int kLim = blockRow + 128;                 // covers block's diagonal

  v8f acc[8] = {};
  block_mma_128x128(Pb, SEQ, Vb + (size_t)colBase * SEQ, SEQ,
                    rowBase, kLim, tile, acc);

  const int hi16 = lane >> 4;
#pragma unroll
  for (int t = 0; t < 8; ++t) {
    const int col = colBase + 16 * t + (lane & 15);
#pragma unroll
    for (int j = 0; j < 8; ++j) {
      const int row = rowBase + j + 8 * hi16;
      Ob[(size_t)row * EMBED + col] = acc[t][j];
    }
  }
}

// -------------------------- host launch ------------------------------------

extern "C" void kernel_launch(void* const* d_in, const int* in_sizes, int n_in,
                              void* d_out, int out_size, void* d_ws, size_t ws_size,
                              hipStream_t stream) {
  (void)in_sizes; (void)n_in; (void)out_size; (void)ws_size;

  const float* x  = (const float*)d_in[0];
  const float* Wq = (const float*)d_in[1];
  const float* bq = (const float*)d_in[2];
  const float* Wk = (const float*)d_in[3];
  const float* bk = (const float*)d_in[4];
  const float* Wv = (const float*)d_in[5];
  const float* bv = (const float*)d_in[6];
  float* out = (float*)d_out;

  // workspace layout (bytes)
  constexpr size_t XBF  = 0;                                   // 16 MB
  constexpr size_t WBF  = XBF  + (size_t)NROWS * EMBED * 2;    //  6 MB
  constexpr size_t QBF  = WBF  + (size_t)3 * EMBED * EMBED * 2;
  constexpr size_t KBF  = QBF  + (size_t)NROWS * EMBED * 2;
  constexpr size_t VTBF = KBF  + (size_t)NROWS * EMBED * 2;
  constexpr size_t SOFF = VTBF + (size_t)NROWS * EMBED * 2;    // 64 MB f32
  constexpr size_t POFF = SOFF + (size_t)NBATCH * SEQ * SEQ * 4;

  char* ws = (char*)d_ws;
  bf16_t* xbf  = (bf16_t*)(ws + XBF);
  bf16_t* wbf  = (bf16_t*)(ws + WBF);
  bf16_t* qbf  = (bf16_t*)(ws + QBF);
  bf16_t* kbf  = (bf16_t*)(ws + KBF);
  bf16_t* vtbf = (bf16_t*)(ws + VTBF);
  float*  S    = (float*) (ws + SOFF);
  bf16_t* P    = (bf16_t*)(ws + POFF);

  {
    int n = NROWS * EMBED;
    convert_f32_bf16<<<(n / 4 + 255) / 256, 256, 0, stream>>>(x, xbf, n);
    int nw = EMBED * EMBED;
    convert_f32_bf16<<<(nw / 4 + 255) / 256, 256, 0, stream>>>(Wq, wbf, nw);
    convert_f32_bf16<<<(nw / 4 + 255) / 256, 256, 0, stream>>>(
        Wk, wbf + (size_t)EMBED * EMBED, nw);
    convert_f32_bf16<<<(nw / 4 + 255) / 256, 256, 0, stream>>>(
        Wv, wbf + (size_t)2 * EMBED * EMBED, nw);
  }

  qkv_gemm_kernel<<<dim3(NROWS / 128, 24), 256, 0, stream>>>(
      xbf, wbf, bq, bk, bv, qbf, kbf, vtbf);

  scores_kernel<<<dim3(SEQ / 128, SEQ / TILE_N, NBATCH), 256, 0, stream>>>(
      qbf, kbf, S);

  softmax_kernel<<<dim3(SEQ, NBATCH), 256, 0, stream>>>(S, P);

  pv_kernel<<<dim3(SEQ / 128, EMBED / TILE_N, NBATCH), 256, 0, stream>>>(
      P, vtbf, out);
}
